// MusicTransformer_14946486190848
// MI455X (gfx1250) — matrix operand order repaired and checked by reading, not compiled, not run on hardware
//
#include <hip/hip_runtime.h>

// ---------------- model constants ----------------
constexpr int cB   = 16;
constexpr int cS   = 512;
constexpr int cL   = 12;
constexpr int cD   = 512;
constexpr int cH   = 8;
constexpr int cDH  = 64;
constexpr int cDFF = 2048;
constexpr int cV   = 631;
constexpr int cMS  = cB * cS;      // 8192 rows of activations
constexpr int cHD  = cH * cDH;     // 512

typedef __bf16 bf16;
typedef __attribute__((ext_vector_type(16))) __bf16 v16bf;
typedef __attribute__((ext_vector_type(8)))  __bf16 v8bf;
typedef __attribute__((ext_vector_type(8)))  float  v8f;

// ---------------- WMMA fragment helpers ----------------
// A-matrix 16x32 bf16 (ISA 7.12.2): lane r = lane&15 holds row r.
// VGPR0-7 hold K = kb..kb+15 (contiguous order chosen here, same as B; not the ISA table order),
// kb = (lane<16) ? 0 : 16  -> two contiguous 16B chunks per lane.
static __device__ inline v16bf load_a16x32(const bf16* base, int ld, int k0, int lane) {
  int r  = lane & 15;
  int kb = (lane >> 4) * 16;
  const bf16* p = base + (size_t)r * ld + k0 + kb;
  v8bf lo = *(const v8bf*)p;
  v8bf hi = *(const v8bf*)(p + 8);
  v16bf a;
#pragma unroll
  for (int i = 0; i < 8; ++i) { a[i] = lo[i]; a[i + 8] = hi[i]; }
  return a;
}

// B-matrix 32x16 bf16: lane holds column n = lane&15; lanes 0-15 K=0..15,
// lanes 16-31 K=16..31, contiguous per lane -> single 32B load from a
// row-major [N,K] ("Bt") buffer.
static __device__ inline v16bf load_b32x16(const bf16* btRow, int k0, int lane) {
  int kk = k0 + (lane >> 4) * 16;
  return *(const v16bf*)(btRow + kk);
}

static __device__ inline v8f wmma_bf16(v16bf a, v16bf b, v8f c) {
  return __builtin_amdgcn_wmma_f32_16x16x32_bf16(false, a, false, b, (short)0, c, false, false);
}

// ---------------- weight conversion ----------------
// Bt[n*K + k] = (bf16) W[k*N + n]   (W is row-major [K,N])
__global__ void cvt_transpose_kernel(const float* __restrict__ W, bf16* __restrict__ Bt,
                                     int K, int N) {
  int idx = blockIdx.x * blockDim.x + threadIdx.x;
  if (idx >= N * K) return;
  int k = idx % K;
  int n = idx / K;
  Bt[idx] = (bf16)W[(size_t)k * N + n];
}

__global__ void cvt_copy_kernel(const float* __restrict__ W, bf16* __restrict__ O, int total) {
  int idx = blockIdx.x * blockDim.x + threadIdx.x;
  if (idx < total) O[idx] = (bf16)W[idx];
}

// ---------------- embedding ----------------
__global__ void embed_kernel(const int* __restrict__ data, const float* __restrict__ emb,
                             const float* __restrict__ pos, float* __restrict__ h,
                             bf16* __restrict__ hb) {
  int idx = blockIdx.x * blockDim.x + threadIdx.x;
  if (idx >= cMS * cD) return;
  int d  = idx & (cD - 1);
  int bs = idx >> 9;            // cD == 512
  int s  = bs & (cS - 1);
  float v = emb[(size_t)data[bs] * cD + d] + pos[(size_t)s * cD + d];
  h[idx]  = v;
  hb[idx] = (bf16)v;
}

// ---------------- generic bf16 WMMA GEMM ----------------
// C[M,N] = A[M,K] @ Bt[N,K]^T (+bias, +relu).
// Wave computes a 32x64 tile: 2 A-frags x 4 B-frags -> 8 WMMAs per K=32 step,
// branch-free inner loop (OOB columns are clamped on load, guarded on store).
__global__ void gemm_bf16_kernel(const bf16* __restrict__ A, const bf16* __restrict__ Bt,
                                 const float* __restrict__ bias,
                                 float* __restrict__ Cf, bf16* __restrict__ Cb,
                                 int M, int N, int K, int relu) {
  int lane  = threadIdx.x & 31;
  int wave  = threadIdx.x >> 5;
  int row0  = (blockIdx.y * 8 + wave) * 32;
  int ncol0 = blockIdx.x * 64;
  if (row0 >= M) return;

  v8f acc0[4], acc1[4];
#pragma unroll
  for (int j = 0; j < 4; ++j)
#pragma unroll
    for (int i = 0; i < 8; ++i) { acc0[j][i] = 0.f; acc1[j][i] = 0.f; }

  const bf16* Abase0 = A + (size_t)row0 * K;
  const bf16* Abase1 = Abase0 + (size_t)16 * K;

  int nlo = lane & 15;
  const bf16* Brow[4];
#pragma unroll
  for (int j = 0; j < 4; ++j) {
    int n = ncol0 + j * 16 + nlo;
    n = (n < N) ? n : (N - 1);          // clamp: no exec-mask dance in K-loop
    Brow[j] = Bt + (size_t)n * K;
  }

  for (int k0 = 0; k0 < K; k0 += 32) {
    v16bf a0  = load_a16x32(Abase0, K, k0, lane);
    v16bf a1  = load_a16x32(Abase1, K, k0, lane);
    v16bf bb0 = load_b32x16(Brow[0], k0, lane);
    v16bf bb1 = load_b32x16(Brow[1], k0, lane);
    v16bf bb2 = load_b32x16(Brow[2], k0, lane);
    v16bf bb3 = load_b32x16(Brow[3], k0, lane);
    acc0[0] = wmma_bf16(a0, bb0, acc0[0]);
    acc1[0] = wmma_bf16(a1, bb0, acc1[0]);
    acc0[1] = wmma_bf16(a0, bb1, acc0[1]);
    acc1[1] = wmma_bf16(a1, bb1, acc1[1]);
    acc0[2] = wmma_bf16(a0, bb2, acc0[2]);
    acc1[2] = wmma_bf16(a1, bb2, acc1[2]);
    acc0[3] = wmma_bf16(a0, bb3, acc0[3]);
    acc1[3] = wmma_bf16(a1, bb3, acc1[3]);
  }

  // C/D layout: lane n = lane&15; rows = (lane>>4)*8 + vgpr
  int r8 = (lane >> 4) * 8;
#pragma unroll
  for (int hm = 0; hm < 2; ++hm) {
    int rbase = row0 + hm * 16 + r8;
#pragma unroll
    for (int j = 0; j < 4; ++j) {
      int n = ncol0 + j * 16 + nlo;
      if (n >= N) continue;
      float bv = bias ? bias[n] : 0.f;
#pragma unroll
      for (int rr = 0; rr < 8; ++rr) {
        float v = (hm ? acc1[j][rr] : acc0[j][rr]) + bv;
        if (relu) v = fmaxf(v, 0.f);
        size_t idx = (size_t)(rbase + rr) * N + n;
        if (Cf) Cf[idx] = v;
        if (Cb) Cb[idx] = (bf16)v;
      }
    }
  }
}

// ---------------- RoPE + layout prep ----------------
// q,k rotary; fold softmax scale (1/8) into q; emit bf16 q,k in [b,s,h,dh]
// and V transposed to [b,h,dh,s] so AV B-fragments are contiguous.
__global__ void rope_kernel(const float* __restrict__ qf, const float* __restrict__ kvf,
                            bf16* __restrict__ qb, bf16* __restrict__ kb,
                            bf16* __restrict__ vT) {
  int idx = blockIdx.x * blockDim.x + threadIdx.x;
  constexpr int total = cB * cS * cH * (cDH / 2);
  if (idx >= total) return;
  int p  = idx % (cDH / 2);
  int t  = idx / (cDH / 2);
  int hh = t % cH;
  int bs = t / cH;
  int s  = bs % cS;
  int b  = bs / cS;

  float ang = (float)s * __powf(10000.0f, -(float)(2 * p) / (float)cDH);
  float sn = __sinf(ang), cs = __cosf(ang);

  size_t qi = (size_t)bs * cHD + hh * cDH + 2 * p;
  float q0 = qf[qi], q1 = qf[qi + 1];
  qb[qi]     = (bf16)((q0 * cs - q1 * sn) * 0.125f);
  qb[qi + 1] = (bf16)((q1 * cs + q0 * sn) * 0.125f);

  size_t ki = (size_t)bs * (2 * cHD) + hh * cDH + 2 * p;
  float k0 = kvf[ki], k1 = kvf[ki + 1];
  size_t ko = (size_t)bs * cHD + hh * cDH + 2 * p;
  kb[ko]     = (bf16)(k0 * cs - k1 * sn);
  kb[ko + 1] = (bf16)(k1 * cs + k0 * sn);

  size_t vi = ki + cHD;
  size_t vo = (((size_t)b * cH + hh) * cDH + 2 * p) * cS + s;
  vT[vo]      = (bf16)kvf[vi];
  vT[vo + cS] = (bf16)kvf[vi + 1];
}

// ---------------- causal attention (LDS-resident scores) ----------------
// 128 threads = 4 waves; wave owns one (b,h,q-tile16). Scores for all keys
// of the tile live in LDS (<= 16x512 f32/wave, ~196KB/block total).
constexpr int SSTR = cS + 8;    // f32 score row stride (bank-conflict pad)
constexpr int PSTR = cS + 16;   // bf16 prob row stride

__global__ void attn_kernel(const bf16* __restrict__ qb, const bf16* __restrict__ kb,
                            const bf16* __restrict__ vT, bf16* __restrict__ av) {
  extern __shared__ char smem[];
  int lane = threadIdx.x & 31;
  int wave = threadIdx.x >> 5;
  int gw = blockIdx.x * 4 + wave;        // over B*H*(S/16) = 4096
  int qt = gw & 31;                      // S/16 == 32
  int bh = gw >> 5;
  int hh = bh % cH;
  int b  = bh / cH;

  float* sc = (float*)smem + (size_t)wave * 16 * SSTR;
  bf16*  pr = (bf16*)(smem + (size_t)4 * 16 * SSTR * sizeof(float)) + (size_t)wave * 16 * PSTR;

  int r = lane & 15, half = lane >> 4;

  const bf16* qp = qb + (((size_t)b * cS + qt * 16) * cH + hh) * cDH;
  v16bf qa0 = load_a16x32(qp, cHD, 0, lane);
  v16bf qa1 = load_a16x32(qp, cHD, 32, lane);

  int nkt = qt + 1;                      // causal: key tiles 0..qt
  for (int kt = 0; kt < nkt; ++kt) {
    const bf16* kp = kb + (((size_t)b * cS + kt * 16) * cH + hh) * cDH;
    // B = K^T: lane's column n = r is key row kt*16+r, contiguous features
    v16bf b0 = *(const v16bf*)(kp + (size_t)r * cHD + half * 16);
    v16bf b1 = *(const v16bf*)(kp + (size_t)r * cHD + 32 + half * 16);
    v8f s8;
#pragma unroll
    for (int i = 0; i < 8; ++i) s8[i] = 0.f;
    s8 = wmma_bf16(qa0, b0, s8);
    s8 = wmma_bf16(qa1, b1, s8);
    int key = kt * 16 + r;
#pragma unroll
    for (int rr = 0; rr < 8; ++rr) {
      int   m = half * 8 + rr;
      float v = s8[rr];
      if (key > qt * 16 + m) v = -3.0e38f;   // causal mask
      sc[m * SSTR + kt * 16 + r] = v;
    }
  }
  asm volatile("s_wait_dscnt 0" ::: "memory");

  // per-row softmax: lane handles row r, two lanes (half) split the keys
  int nk = nkt * 16;
  float mx = -3.0e38f;
  for (int kk = half; kk < nk; kk += 2) mx = fmaxf(mx, sc[r * SSTR + kk]);
  mx = fmaxf(mx, __shfl_xor(mx, 16, 32));
  float sum = 0.f;
  for (int kk = half; kk < nk; kk += 2) sum += __expf(sc[r * SSTR + kk] - mx);
  sum += __shfl_xor(sum, 16, 32);
  float inv = 1.0f / sum;
  for (int kk = half; kk < nk; kk += 2)
    pr[r * PSTR + kk] = (bf16)(__expf(sc[r * SSTR + kk] - mx) * inv);
  int nk32 = (nk + 31) & ~31;
  for (int kk = nk + half; kk < nk32; kk += 2) pr[r * PSTR + kk] = (bf16)0.f;
  asm volatile("s_wait_dscnt 0" ::: "memory");

  // AV: probs re-read from LDS in A-fragment layout (LDS does the transpose)
  v8f acc[4];
#pragma unroll
  for (int c = 0; c < 4; ++c)
#pragma unroll
    for (int i = 0; i < 8; ++i) acc[c][i] = 0.f;

  const bf16* vp = vT + ((size_t)b * cH + hh) * cDH * cS;
  for (int t = 0; t < nk32 / 32; ++t) {
    const bf16* pp = pr + r * PSTR + t * 32 + half * 16;
    v8bf lo = *(const v8bf*)pp;
    v8bf hi = *(const v8bf*)(pp + 8);
    v16bf a;
#pragma unroll
    for (int i = 0; i < 8; ++i) { a[i] = lo[i]; a[i + 8] = hi[i]; }
#pragma unroll
    for (int c = 0; c < 4; ++c) {
      v16bf bb = *(const v16bf*)(vp + (size_t)(c * 16 + r) * cS + t * 32 + half * 16);
      acc[c] = wmma_bf16(a, bb, acc[c]);
    }
  }

  bf16* op = av + (((size_t)b * cS + qt * 16) * cH + hh) * cDH;
#pragma unroll
  for (int c = 0; c < 4; ++c)
#pragma unroll
    for (int rr = 0; rr < 8; ++rr) {
      int m = half * 8 + rr;
      op[(size_t)m * cHD + c * 16 + r] = (bf16)acc[c][rr];
    }
}

// ---------------- residual + LayerNorm ----------------
__global__ void ln_residual_kernel(const float* __restrict__ hin, const float* __restrict__ delta,
                                   const float* __restrict__ g, const float* __restrict__ beta,
                                   float* __restrict__ hout, bf16* __restrict__ hbout) {
  int row  = blockIdx.x * 8 + (threadIdx.x >> 5);
  int lane = threadIdx.x & 31;
  const float* hp = hin   + (size_t)row * cD;
  const float* dp = delta + (size_t)row * cD;
  float x[16];
  float s = 0.f;
#pragma unroll
  for (int i = 0; i < 16; ++i) {
    float v = hp[i * 32 + lane] + dp[i * 32 + lane];
    x[i] = v; s += v;
  }
#pragma unroll
  for (int o = 16; o > 0; o >>= 1) s += __shfl_xor(s, o, 32);
  float mean = s * (1.0f / cD);
  float var = 0.f;
#pragma unroll
  for (int i = 0; i < 16; ++i) { float d = x[i] - mean; var += d * d; }
#pragma unroll
  for (int o = 16; o > 0; o >>= 1) var += __shfl_xor(var, o, 32);
  float rstd = rsqrtf(var * (1.0f / cD) + 1e-5f);
#pragma unroll
  for (int i = 0; i < 16; ++i) {
    int c = i * 32 + lane;
    float y = (x[i] - mean) * rstd * g[c] + beta[c];
    hout[(size_t)row * cD + c]  = y;
    hbout[(size_t)row * cD + c] = (bf16)y;
  }
}

// ---------------- host orchestration ----------------
extern "C" void kernel_launch(void* const* d_in, const int* in_sizes, int n_in,
                              void* d_out, int out_size, void* d_ws, size_t ws_size,
                              hipStream_t stream) {
  (void)in_sizes; (void)n_in; (void)out_size; (void)ws_size;

  const int*   data = (const int*)d_in[0];
  const float* emb  = (const float*)d_in[1];
  const float* pos  = (const float*)d_in[2];
  const float* Wq   = (const float*)d_in[3];
  const float* Wkv  = (const float*)d_in[4];
  const float* Wo   = (const float*)d_in[5];
  const float* W1   = (const float*)d_in[6];
  const float* b1   = (const float*)d_in[7];
  const float* W2   = (const float*)d_in[8];
  const float* b2   = (const float*)d_in[9];
  const float* ln1g = (const float*)d_in[10];
  const float* ln1b = (const float*)d_in[11];
  const float* ln2g = (const float*)d_in[12];
  const float* ln2b = (const float*)d_in[13];
  const float* outb = (const float*)d_in[14];

  char*  ws  = (char*)d_ws;
  size_t off = 0;
  auto carve = [&](size_t bytes) -> char* {
    char* p = ws + off;
    off += (bytes + 255) & ~(size_t)255;
    return p;
  };

  float* h    = (float*)carve((size_t)cMS * cD * 4);
  bf16*  hb   = (bf16*) carve((size_t)cMS * cD * 2);
  float* tmp  = (float*)carve((size_t)cMS * cD * 4);
  float* qf   = (float*)carve((size_t)cMS * cHD * 4);
  float* kvf  = (float*)carve((size_t)cMS * 2 * cHD * 4);
  bf16*  qbf  = (bf16*) carve((size_t)cMS * cHD * 2);
  bf16*  kbf  = (bf16*) carve((size_t)cMS * cHD * 2);
  bf16*  vT   = (bf16*) carve((size_t)cB * cH * cDH * cS * 2);
  bf16*  avb  = (bf16*) carve((size_t)cMS * cHD * 2);
  bf16*  ff1  = (bf16*) carve((size_t)cMS * cDFF * 2);
  bf16*  Wqt  = (bf16*) carve((size_t)cL * cHD * cD * 2);
  bf16*  Wkvt = (bf16*) carve((size_t)cL * 2 * cHD * cD * 2);
  bf16*  Wot  = (bf16*) carve((size_t)cL * cD * cHD * 2);
  bf16*  W1t  = (bf16*) carve((size_t)cL * cDFF * cD * 2);
  bf16*  W2t  = (bf16*) carve((size_t)cL * cD * cDFF * 2);
  bf16*  embb = (bf16*) carve((size_t)cV * cD * 2);

  auto cvtT = [&](const float* W, bf16* Bt, int K, int N) {
    int total = N * K;
    cvt_transpose_kernel<<<(total + 255) / 256, 256, 0, stream>>>(W, Bt, K, N);
  };

  // one-time (per launch) weight prep: bf16 + [N,K] transpose
  for (int l = 0; l < cL; ++l) {
    cvtT(Wq  + (size_t)l * cD * cHD,      Wqt  + (size_t)l * cHD * cD,      cD,   cHD);
    cvtT(Wkv + (size_t)l * cD * 2 * cHD,  Wkvt + (size_t)l * 2 * cHD * cD,  cD,   2 * cHD);
    cvtT(Wo  + (size_t)l * cHD * cD,      Wot  + (size_t)l * cD * cHD,      cHD,  cD);
    cvtT(W1  + (size_t)l * cD * cDFF,     W1t  + (size_t)l * cDFF * cD,     cD,   cDFF);
    cvtT(W2  + (size_t)l * cDFF * cD,     W2t  + (size_t)l * cD * cDFF,     cDFF, cD);
  }
  // emb is already [V, D] = [N, K] for the logits GEMM (B = emb^T)
  cvt_copy_kernel<<<(cV * cD + 255) / 256, 256, 0, stream>>>(emb, embb, cV * cD);

  embed_kernel<<<(cMS * cD + 255) / 256, 256, 0, stream>>>(data, emb, pos, h, hb);

  auto gemm = [&](const bf16* A, const bf16* Bt, const float* bias,
                  float* Cf, bf16* Cb, int M, int N, int K, int relu) {
    dim3 g((N + 63) / 64, (M + 255) / 256);   // 8 waves x 32 rows per block
    gemm_bf16_kernel<<<g, 256, 0, stream>>>(A, Bt, bias, Cf, Cb, M, N, K, relu);
  };

  constexpr int ropeThreads = cB * cS * cH * (cDH / 2);
  constexpr size_t attnLds = (size_t)4 * (16 * SSTR * 4 + 16 * PSTR * 2);
  const int attnBlocks = (cB * cH * (cS / 16)) / 4;   // 1024
  const int lnBlocks   = cMS / 8;                     // 1024

  for (int l = 0; l < cL; ++l) {
    // projections
    gemm(hb, Wqt  + (size_t)l * cHD * cD,     nullptr, qf,  nullptr, cMS, cHD,     cD, 0);
    gemm(hb, Wkvt + (size_t)l * 2 * cHD * cD, nullptr, kvf, nullptr, cMS, 2 * cHD, cD, 0);
    // RoPE + scale fold + V transpose
    rope_kernel<<<(ropeThreads + 255) / 256, 256, 0, stream>>>(qf, kvf, qbf, kbf, vT);
    // attention
    attn_kernel<<<attnBlocks, 128, attnLds, stream>>>(qbf, kbf, vT, avb);
    // output projection + residual + LN1
    gemm(avb, Wot + (size_t)l * cD * cHD, nullptr, tmp, nullptr, cMS, cD, cHD, 0);
    ln_residual_kernel<<<lnBlocks, 256, 0, stream>>>(h, tmp, ln1g + (size_t)l * cD,
                                                     ln1b + (size_t)l * cD, h, hb);
    // FFN
    gemm(hb,  W1t + (size_t)l * cDFF * cD, b1 + (size_t)l * cDFF, nullptr, ff1, cMS, cDFF, cD,   1);
    gemm(ff1, W2t + (size_t)l * cD * cDFF, b2 + (size_t)l * cD,   tmp,    nullptr, cMS, cD, cDFF, 0);
    ln_residual_kernel<<<lnBlocks, 256, 0, stream>>>(h, tmp, ln2g + (size_t)l * cD,
                                                     ln2b + (size_t)l * cD, h, hb);
  }

  // logits: C = h @ emb^T + out_b  (Bt = emb itself, N = 631 guarded)
  gemm(hb, embb, outb, (float*)d_out, nullptr, cMS, cV, cD, 0);
}
